// intentExtractor_54219667145024
// MI455X (gfx1250) — matrix operand order repaired and checked by reading, not compiled, not run on hardware
//
#include <hip/hip_runtime.h>

#define N_H 4
#define N_I 32
#define N_B 8
#define D_MODEL 256
#define D_K 64
#define N_IS 288
#define BS_ 256
#define MAXLEN_ 200
#define POSP 208
#define NTILE_POS 13
#define NTILE_Q 18

typedef __attribute__((ext_vector_type(16))) __bf16 v16bf;
typedef __attribute__((ext_vector_type(8)))  __bf16 v8bf;
typedef __attribute__((ext_vector_type(8)))  float  v8f;

struct B32 { uint4 a, b; };
struct H8  { unsigned short h[8]; };

__device__ __forceinline__ v16bf ld16(const __bf16* p, size_t stride) {
  v16bf v;
  #pragma unroll
  for (int s = 0; s < 16; ++s) v[s] = p[(size_t)(s < 8 ? s : s + 8) * stride];
  return v;
}

// A-fragment (16x32 bf16): lane<16 holds row (lane&15), K={kb+0..7, kb+16..23};
// lane>=16 same row, K={kb+8..15, kb+24..31}
__device__ __forceinline__ v16bf ld_a_frag(const __bf16* rowp, int kbase, int loHalf) {
  B32 t;
  t.a = *reinterpret_cast<const uint4*>(rowp + kbase + loHalf);
  t.b = *reinterpret_cast<const uint4*>(rowp + kbase + 16 + loHalf);
  return __builtin_bit_cast(v16bf, t);
}

__device__ __forceinline__ v8f wmma_bf16(v16bf a, v16bf b, v8f c) {
  return __builtin_amdgcn_wmma_f32_16x16x32_bf16(false, a, false, b, (short)0, c, false, false);
}

// _get_cn: integer-valued float in [0, 11]
__device__ __forceinline__ float get_cn(float xf) {
  const float quarter = 6.0f;                 // NB/4
  if (xf < quarter) return xf;
  float f1 = quarter + floorf(logf(xf / quarter) / logf(100.0f / 3.0f) * quarter);
  return fminf(f1, 11.0f);                    // NB/2 - 1
}

__global__ void cvt_bf16(const float* __restrict__ src, unsigned short* __restrict__ dst, int n) {
  int i = blockIdx.x * blockDim.x + threadIdx.x;
  if (i < n) dst[i] = __builtin_bit_cast(unsigned short, (__bf16)src[i]);
}

// ---------------------------------------------------------------------------
// Projections of item -> K/V (type-gathered Bs + shared Ba), WMMA bf16.
// One block = (b, 16-position tile). 8 waves, 64 output 16x16 tiles
// (4 outputs x 16 n-tiles). K stored transposed [b,h,d,POSP]; V [b,h,POSP,64].
// ---------------------------------------------------------------------------
__global__ void __launch_bounds__(256) proj_kv(
    const unsigned short* __restrict__ item_u, const unsigned short* __restrict__ Witem_u,
    const int* __restrict__ b_seq,
    unsigned short* __restrict__ KBsT_u, unsigned short* __restrict__ VBs_u,
    unsigned short* __restrict__ KBaT_u, unsigned short* __restrict__ VBa_u) {
  const __bf16* item  = reinterpret_cast<const __bf16*>(item_u);
  const __bf16* Witem = reinterpret_cast<const __bf16*>(Witem_u);
  int blk = blockIdx.x;
  int b = blk / NTILE_POS, tile = blk % NTILE_POS;
  int tid = threadIdx.x, wave = tid >> 5, lane = tid & 31;

  __shared__ __bf16 sA[16 * 256];
  __shared__ int sType[16];
  {
    int r = tid >> 4, c = (tid & 15) * 16;
    int pos = tile * 16 + r;
    uint4 lo = {0, 0, 0, 0}, hi = {0, 0, 0, 0};
    if (pos < MAXLEN_) {
      const __bf16* src = item + ((size_t)b * MAXLEN_ + pos) * 256 + c;
      lo = *reinterpret_cast<const uint4*>(src);
      hi = *reinterpret_cast<const uint4*>(src + 8);
    }
    *reinterpret_cast<uint4*>(&sA[r * 256 + c])     = lo;
    *reinterpret_cast<uint4*>(&sA[r * 256 + c + 8]) = hi;
    if (tid < 16) {
      int p2 = tile * 16 + tid;
      sType[tid] = (p2 < MAXLEN_) ? b_seq[b * MAXLEN_ + p2] : -1;
    }
  }
  __syncthreads();

  int row = lane & 15;
  int loHalf = (lane < 16) ? 0 : 8;
  int myType = sType[row];
  const __bf16* Arow = &sA[row * 256];

  for (int i = 0; i < 8; ++i) {
    int id = i * 8 + wave;            // interleave outputs across waves
    int outm = id >> 4, ntile = id & 15;
    int sel = outm & 1;               // 0 = Key, 1 = Value
    v8f acc = {};
    if (outm < 2) {                   // Bs: per-row weight type, masked passes
      for (int t = 0; t < N_B + 2; ++t) {
        if (!__any((int)(myType == t))) continue;   // wave-uniform skip
        bool on = (myType == t);
        const __bf16* W = Witem + (((size_t)sel * 10 + t) * 256) * 256 + ntile * 16;
        #pragma unroll
        for (int kc = 0; kc < 8; ++kc) {
          v16bf a = ld_a_frag(Arow, kc * 32, loHalf);
          v16bf az = {};
          a = on ? a : az;            // zero rows of other types (EXEC stays full)
          v16bf bf = ld16(W + (size_t)(kc * 32 + loHalf) * 256 + (lane & 15), 256);
          acc = wmma_bf16(a, bf, acc);
        }
      }
    } else {                          // Ba: shared weight (last type slot)
      bool on = (myType >= 0);        // zero pad rows
      const __bf16* W = Witem + (((size_t)sel * 10 + (N_B + 1)) * 256) * 256 + ntile * 16;
      #pragma unroll
      for (int kc = 0; kc < 8; ++kc) {
        v16bf a = ld_a_frag(Arow, kc * 32, loHalf);
        v16bf az = {};
        a = on ? a : az;
        v16bf bf = ld16(W + (size_t)(kc * 32 + loHalf) * 256 + (lane & 15), 256);
        acc = wmma_bf16(a, bf, acc);
      }
    }
    // C/D layout: VGPR r, lanes0-15 -> (M=r, N=lane); lanes16-31 -> (M=r+8, N=lane-16)
    int hk = ntile * 16 + (lane & 15);
    int h = hk >> 6, d = hk & 63;
    int posb = tile * 16 + ((lane < 16) ? 0 : 8);
    if (sel == 0) {
      __bf16* KT = reinterpret_cast<__bf16*>((outm == 0) ? KBsT_u : KBaT_u);
      H8 pk;
      #pragma unroll
      for (int r = 0; r < 8; ++r) pk.h[r] = __builtin_bit_cast(unsigned short, (__bf16)acc[r]);
      *reinterpret_cast<uint4*>(KT + (((size_t)(b * 4 + h) * 64 + d) * POSP + posb)) =
          __builtin_bit_cast(uint4, pk);
    } else {
      __bf16* Vp = reinterpret_cast<__bf16*>((outm == 1) ? VBs_u : VBa_u);
      __bf16* dst = Vp + (((size_t)(b * 4 + h) * POSP + posb) * 64 + d);
      #pragma unroll
      for (int r = 0; r < 8; ++r) dst[(size_t)r * 64] = (__bf16)acc[r];
    }
  }
}

// ---------------------------------------------------------------------------
// tQuery: intent (shared rows) x W_intent[b_seq2[b,q]] -> Q[b,h,q,64] bf16
// ---------------------------------------------------------------------------
__global__ void __launch_bounds__(256) proj_q(
    const unsigned short* __restrict__ intent_u, const unsigned short* __restrict__ Wint_u,
    const int* __restrict__ b_seq2, unsigned short* __restrict__ Q_u) {
  const __bf16* intent = reinterpret_cast<const __bf16*>(intent_u);
  const __bf16* Wint   = reinterpret_cast<const __bf16*>(Wint_u);
  __bf16* Q = reinterpret_cast<__bf16*>(Q_u);
  int blk = blockIdx.x;
  int b = blk / NTILE_Q, tile = blk % NTILE_Q;
  int tid = threadIdx.x, wave = tid >> 5, lane = tid & 31;

  __shared__ __bf16 sA[16 * 256];
  __shared__ int sType[16];
  {
    int r = tid >> 4, c = (tid & 15) * 16;
    int q = tile * 16 + r;                       // 288 rows exactly, no pad
    const __bf16* src = intent + (size_t)q * 256 + c;
    *reinterpret_cast<uint4*>(&sA[r * 256 + c])     = *reinterpret_cast<const uint4*>(src);
    *reinterpret_cast<uint4*>(&sA[r * 256 + c + 8]) = *reinterpret_cast<const uint4*>(src + 8);
    if (tid < 16) sType[tid] = b_seq2[b * N_IS + tile * 16 + tid];
  }
  __syncthreads();

  int row = lane & 15, loHalf = (lane < 16) ? 0 : 8;
  int myType = sType[row];
  const __bf16* Arow = &sA[row * 256];

  for (int i = 0; i < 2; ++i) {
    int ntile = i * 8 + wave;
    v8f acc = {};
    for (int t = 0; t < N_B + 1; ++t) {
      if (!__any((int)(myType == t))) continue;
      bool on = (myType == t);
      const __bf16* W = Wint + ((size_t)t * 256) * 256 + ntile * 16;
      #pragma unroll
      for (int kc = 0; kc < 8; ++kc) {
        v16bf a = ld_a_frag(Arow, kc * 32, loHalf);
        v16bf az = {};
        a = on ? a : az;
        v16bf bf = ld16(W + (size_t)(kc * 32 + loHalf) * 256 + (lane & 15), 256);
        acc = wmma_bf16(a, bf, acc);
      }
    }
    int hk = ntile * 16 + (lane & 15);
    int h = hk >> 6, d = hk & 63;
    int qb = tile * 16 + ((lane < 16) ? 0 : 8);
    __bf16* dst = Q + (((size_t)(b * 4 + h) * N_IS + qb) * 64 + d);
    #pragma unroll
    for (int r = 0; r < 8; ++r) dst[(size_t)r * 64] = (__bf16)acc[r];
  }
}

// ---------------------------------------------------------------------------
// Attention: 1 wave per (b, h, 16-query tile). WMMA scores -> LDS ->
// mask/softmax/top-k (k<=11, exact stable ranks) -> sparse p.V -> f32 out.
// ---------------------------------------------------------------------------
__global__ void __launch_bounds__(32) attn_kernel(
    const unsigned short* __restrict__ Q_u,
    const unsigned short* __restrict__ KBsT_u, const unsigned short* __restrict__ VBs_u,
    const unsigned short* __restrict__ KBaT_u, const unsigned short* __restrict__ VBa_u,
    const int* __restrict__ maskp, const int* __restrict__ type_cnt,
    float* __restrict__ outp) {
  int blk = blockIdx.x;
  int qt = blk % NTILE_Q;
  int h  = (blk / NTILE_Q) & 3;
  int b  = blk / (NTILE_Q * 4);
  int lane = threadIdx.x;
  bool isBa = (qt >= 16);
  const __bf16* KT = reinterpret_cast<const __bf16*>(isBa ? KBaT_u : KBsT_u);
  const __bf16* V  = reinterpret_cast<const __bf16*>(isBa ? VBa_u  : VBs_u);
  const __bf16* Q  = reinterpret_cast<const __bf16*>(Q_u);
  size_t bh = (size_t)b * 4 + h;

  __shared__ float sc[16 * POSP];

  int q0 = qt * 16;
  int loHalf = (lane < 16) ? 0 : 8;
  const __bf16* qptr = Q + (bh * N_IS + q0 + (lane & 15)) * 64;
  v16bf aq0 = ld_a_frag(qptr, 0,  loHalf);   // d 0..31
  v16bf aq1 = ld_a_frag(qptr, 32, loHalf);   // d 32..63

  int mbase = (lane < 16) ? 0 : 8;
  int kcol  = lane & 15;
  for (int kt = 0; kt < NTILE_POS; ++kt) {
    const __bf16* kb = KT + (bh * 64) * POSP + kt * 16;
    v8f s = {};
    v16bf b0 = ld16(kb + (size_t)loHalf * POSP + (lane & 15), POSP);          // B col pos=lane&15, d 0..31
    s = wmma_bf16(aq0, b0, s);
    v16bf b1 = ld16(kb + (size_t)(32 + loHalf) * POSP + (lane & 15), POSP);   // B col pos=lane&15, d 32..63
    s = wmma_bf16(aq1, b1, s);
    #pragma unroll
    for (int r = 0; r < 8; ++r) sc[(mbase + r) * POSP + kt * 16 + kcol] = s[r];
  }
  __syncthreads();

  if (lane < 16) {
    int q = q0 + lane;
    int kk;
    if (q < N_B * N_I) {
      kk = (int)get_cn((float)type_cnt[b * (N_B + 1) + 1 + (q >> 5)]);
    } else {
      int ssum = 0;
      for (int j = 0; j < N_B + 1; ++j) ssum += type_cnt[b * (N_B + 1) + j];
      kk = (int)get_cn((float)ssum);
    }
    const int* mrow = maskp + ((size_t)b * N_IS + q) * MAXLEN_;
    float* row = &sc[lane * POSP];

    float mx = -3.4e38f;
    for (int j = 0; j < POSP; ++j) {
      float v;
      if (j < MAXLEN_) v = (mrow[j] != 0) ? row[j] * 0.125f : -1e30f;  // 1/sqrt(64)
      else             v = -__builtin_inff();                           // pad keys
      row[j] = v;
      mx = fmaxf(mx, v);
    }
    float Z = 0.0f;
    for (int j = 0; j < POSP; ++j) {
      float p = __expf(row[j] - mx);
      row[j] = p;
      Z += p;
    }
    float inv = 1.0f / Z;

    float o[64];
    #pragma unroll
    for (int d = 0; d < 64; ++d) o[d] = 0.0f;

    for (int it = 0; it < kk; ++it) {          // kk <= 11
      float best = -1.0f; int bj = 0;
      for (int j = 0; j < POSP; ++j) {         // strict > keeps smallest index on ties
        float p = row[j];
        if (p > best) { best = p; bj = j; }
      }
      row[bj] = -2.0f;                          // remove from candidates
      float w = best * inv;
      const __bf16* vrow = V + (bh * POSP + bj) * 64;
      #pragma unroll
      for (int c = 0; c < 8; ++c) {
        uint4 u = *reinterpret_cast<const uint4*>(vrow + c * 8);
        v8bf vv = __builtin_bit_cast(v8bf, u);
        #pragma unroll
        for (int e = 0; e < 8; ++e) o[c * 8 + e] += w * (float)vv[e];
      }
    }
    float* op = outp + ((size_t)b * N_IS + q) * (N_H * D_K) + h * D_K;
    #pragma unroll
    for (int d = 0; d < 64; ++d) op[d] = o[d];
  }
}

// ---------------------------------------------------------------------------
extern "C" void kernel_launch(void* const* d_in, const int* in_sizes, int n_in,
                              void* d_out, int out_size, void* d_ws, size_t ws_size,
                              hipStream_t stream) {
  const float* item     = (const float*)d_in[0];
  const float* intent   = (const float*)d_in[1];
  const int*   maskp    = (const int*)d_in[2];
  const int*   b_seq    = (const int*)d_in[3];
  const int*   b_seq2   = (const int*)d_in[4];
  const int*   type_cnt = (const int*)d_in[5];
  const float* W_item   = (const float*)d_in[6];
  const float* W_intent = (const float*)d_in[7];
  float* outp = (float*)d_out;

  unsigned short* ws = (unsigned short*)d_ws;
  size_t o = 0;
  auto carve = [&](size_t elems) {
    unsigned short* p = ws + o;
    o += (elems + 127) & ~(size_t)127;   // 256-byte aligned carve
    return p;
  };
  unsigned short* item_bf   = carve((size_t)BS_ * MAXLEN_ * 256);
  unsigned short* Witem_bf  = carve((size_t)2 * 10 * 256 * 256);
  unsigned short* Wint_bf   = carve((size_t)9 * 256 * 256);
  unsigned short* intent_bf = carve((size_t)N_IS * 256);
  unsigned short* KBsT      = carve((size_t)BS_ * 4 * 64 * POSP);
  unsigned short* VBs       = carve((size_t)BS_ * 4 * POSP * 64);
  unsigned short* KBaT      = carve((size_t)BS_ * 4 * 64 * POSP);
  unsigned short* VBa       = carve((size_t)BS_ * 4 * POSP * 64);
  unsigned short* Qb        = carve((size_t)BS_ * 4 * N_IS * 64);

  int n_item = BS_ * MAXLEN_ * 256;
  int n_wi   = 2 * 10 * 256 * 256;
  int n_wq   = 9 * 256 * 256;
  int n_int  = N_IS * 256;
  cvt_bf16<<<(n_item + 255) / 256, 256, 0, stream>>>(item, item_bf, n_item);
  cvt_bf16<<<(n_wi   + 255) / 256, 256, 0, stream>>>(W_item, Witem_bf, n_wi);
  cvt_bf16<<<(n_wq   + 255) / 256, 256, 0, stream>>>(W_intent, Wint_bf, n_wq);
  cvt_bf16<<<(n_int  + 255) / 256, 256, 0, stream>>>(intent, intent_bf, n_int);

  proj_kv<<<BS_ * NTILE_POS, 256, 0, stream>>>(item_bf, Witem_bf, b_seq, KBsT, VBs, KBaT, VBa);
  proj_q <<<BS_ * NTILE_Q,   256, 0, stream>>>(intent_bf, Wint_bf, b_seq2, Qb);
  attn_kernel<<<BS_ * 4 * NTILE_Q, 32, 0, stream>>>(Qb, KBsT, VBs, KBaT, VBa, maskp, type_cnt, outp);

  (void)in_sizes; (void)n_in; (void)out_size; (void)ws_size;
}
